// attention_85830626443598
// MI455X (gfx1250) — matrix operand order repaired and checked by reading, not compiled, not run on hardware
//
#include <hip/hip_runtime.h>

#define BDIM 8
#define TDIM 2048
#define DDIM 1024

typedef __attribute__((ext_vector_type(16))) __bf16 v16bf;
typedef __attribute__((ext_vector_type(8)))  __bf16 v8bf;
typedef __attribute__((ext_vector_type(4)))  __bf16 v4bf;
typedef __attribute__((ext_vector_type(8)))  float  v8f;
typedef __attribute__((ext_vector_type(4)))  int    v4i;

// gfx1250 async global->LDS path (guarded: falls back to direct loads)
#if defined(__HIP_DEVICE_COMPILE__) && defined(__has_builtin)
#if __has_builtin(__builtin_amdgcn_global_load_async_to_lds_b128) && \
    __has_builtin(__builtin_amdgcn_s_wait_asynccnt)
#define ASYNC_LDS 1
#endif
#endif

static __device__ __forceinline__ v16bf cat8(v8bf lo, v8bf hi) {
  return __builtin_shufflevector(lo, hi, 0,1,2,3,4,5,6,7,8,9,10,11,12,13,14,15);
}
static __device__ __forceinline__ v8f wmma_bf16f32(v16bf a, v16bf b, v8f c) {
  // 8 args: (neg_a, A, neg_b, B, c_mod, C, reuse_a, reuse_b)
  return __builtin_amdgcn_wmma_f32_16x16x32_bf16(false, a, false, b, (short)0, c,
                                                 false, false);
}
#ifdef ASYNC_LDS
static __device__ __forceinline__ void async_cp16(const void* g, void* l) {
  // builtin signature (from hipcc diagnostic): (v4i AS(1)*, v4i AS(3)*, Ii, Ii)
  __builtin_amdgcn_global_load_async_to_lds_b128(
      (__attribute__((address_space(1))) v4i*)g,
      (__attribute__((address_space(3))) v4i*)l, 0, 0);
}
#endif

// ---------------- kernel 1: f32 -> bf16 convert ----------------
__global__ void cvt_f32_bf16(const float* __restrict__ src,
                             __bf16* __restrict__ dst, int n4) {
  int i = blockIdx.x * blockDim.x + threadIdx.x;
  if (i >= n4) return;
  float4 v = ((const float4*)src)[i];
  v4bf o;
  o[0] = (__bf16)v.x; o[1] = (__bf16)v.y; o[2] = (__bf16)v.z; o[3] = (__bf16)v.w;
  ((v4bf*)dst)[i] = o;
}

// ---------------- kernel 2: QKV projection GEMMs ----------------
// grid = B*T/16 row-tiles, block = 256 (8 waves). Wave w owns output cols
// [w*128, w*128+128). kt-outer / nt-inner: A fragment loaded once per kt and
// reused by 8 independent WMMA accumulator chains (2.25 loads per WMMA).
__global__ __launch_bounds__(256) void qkv_kernel(
    const __bf16* __restrict__ ebf, const __bf16* __restrict__ rbf,
    const __bf16* __restrict__ wq, const __bf16* __restrict__ wk,
    const __bf16* __restrict__ wv,
    const float* __restrict__ bq, const float* __restrict__ bk,
    const float* __restrict__ bv,
    __bf16* __restrict__ Qbf, __bf16* __restrict__ Kbf,
    __bf16* __restrict__ Vt) {
  const int lane = threadIdx.x & 31;
  const int wave = threadIdx.x >> 5;
  const int m0   = blockIdx.x * 16;      // global row in [0, B*T)
  const int bIdx = m0 / TDIM;
  const int tloc = m0 % TDIM;

  const int aM   = lane & 15;            // A: M = lane%16
  const int aKlo = (lane < 16) ? 0 : 8;  // A: K half select
  const int bN   = lane & 15;            // B: N = lane%16
  const int bKlo = (lane < 16) ? 0 : 8;  // B: K half select
  const int cN   = lane & 15;            // C: N = lane%16
  const int cM0  = (lane < 16) ? 0 : 8;  // C: M base

  const __bf16* Xs[3] = { ebf, rbf, ebf };
  const __bf16* Ws[3] = { wq, wk, wv };
  const float*  Bs[3] = { bq, bk, bv };
  __bf16* Ds[2] = { Qbf, Kbf };

  for (int outi = 0; outi < 3; ++outi) {
    const __bf16* X    = Xs[outi];
    const __bf16* W    = Ws[outi];
    const float*  bias = Bs[outi];
    const __bf16* arow = X + (size_t)(m0 + aM) * DDIM + aKlo;

    v8f acc[8];
#pragma unroll
    for (int nt = 0; nt < 8; ++nt) {
      const float bv0 = bias[wave * 128 + nt * 16 + cN];  // bias per out column
#pragma unroll
      for (int r = 0; r < 8; ++r) acc[nt][r] = bv0;
    }

#pragma unroll 2
    for (int kt = 0; kt < 32; ++kt) {
      const __bf16* ap = arow + kt * 32;
      v16bf a = cat8(*(const v8bf*)(ap), *(const v8bf*)(ap + 16));
#pragma unroll
      for (int nt = 0; nt < 8; ++nt) {
        const __bf16* bp =
            W + (size_t)(wave * 128 + nt * 16 + bN) * DDIM + kt * 32 + bKlo;
        v16bf bb = cat8(*(const v8bf*)(bp), *(const v8bf*)(bp + 16));
        acc[nt] = wmma_bf16f32(a, bb, acc[nt]);
      }
    }

#pragma unroll
    for (int nt = 0; nt < 8; ++nt) {
      const int n0 = wave * 128 + nt * 16;
      if (outi < 2) {
        __bf16* Dst = Ds[outi];
#pragma unroll
        for (int r = 0; r < 8; ++r)
          Dst[(size_t)(m0 + cM0 + r) * DDIM + n0 + cN] = (__bf16)acc[nt][r];
      } else {
        // V stored transposed: lane column d fixed, 8 consecutive t -> 1x b128
        v8bf pv;
#pragma unroll
        for (int r = 0; r < 8; ++r) pv[r] = (__bf16)acc[nt][r];
        size_t idx = ((size_t)bIdx * DDIM + n0 + cN) * TDIM + tloc + cM0;
        *(v8bf*)(Vt + idx) = pv;
      }
    }
  }
}

// ---------------- kernel 3: Ksum row reduction ----------------
__global__ __launch_bounds__(256) void ksum_kernel(const __bf16* __restrict__ Kbf,
                                                   float* __restrict__ Ksum) {
  const int lane = threadIdx.x & 31;
  const int row  = blockIdx.x * 8 + (threadIdx.x >> 5);
  const __bf16* p = Kbf + (size_t)row * DDIM + lane * 8;
  float s = 0.f;
#pragma unroll
  for (int i = 0; i < 4; ++i) {
    v8bf v = *(const v8bf*)(p + i * 256);
#pragma unroll
    for (int j = 0; j < 8; ++j) s += (float)v[j];
  }
#pragma unroll
  for (int off = 16; off > 0; off >>= 1) s += __shfl_xor(s, off, 32);
  if (lane == 0) Ksum[row] = s;
}

// ---------------- kernel 4: fused causal (QK^T + bias) @ V ----------------
// grid = B*T/16, block = 256. Wave w owns d-slice [w*128, w*128+128).
// K tiles (contiguous 64KB per 32-s block) are prefetched one block ahead
// into double-buffered LDS via GLOBAL_LOAD_ASYNC_TO_LDS_B128 (ASYNCcnt),
// then consumed through ds_load_b128. Vt stays on the VMEM pipe (per-wave
// disjoint d-slices: LDS staging would save no traffic).
// Dynamic LDS layout: [0,2048) Ssh f32 | [2048,3072) Abf bf16 | [3072,..) Kbuf x2
__global__ __launch_bounds__(256) void attn_kernel(
    const __bf16* __restrict__ Qbf, const __bf16* __restrict__ Kbf,
    const __bf16* __restrict__ Vt,  const float* __restrict__ Ksum,
    const float* __restrict__ vtab, float* __restrict__ out) {
  extern __shared__ __align__(16) char smem_raw[];
  float*  Ssh  = (float*)smem_raw;                 // 16x32 f32
  __bf16* Abf  = (__bf16*)(smem_raw + 2048);       // 16x32 bf16
#ifdef ASYNC_LDS
  __bf16* Klds = (__bf16*)(smem_raw + 3072);       // 2 x (32 x 1024) bf16
#endif

  const int tid  = threadIdx.x;
  const int lane = tid & 31;
  const int wave = tid >> 5;
  const int m0   = blockIdx.x * 16;
  const int bIdx = m0 / TDIM;
  const int t0   = m0 % TDIM;

  const int aM   = lane & 15;
  const int aKlo = (lane < 16) ? 0 : 8;
  const int bN   = lane & 15;
  const int bKlo = (lane < 16) ? 0 : 8;
  const int cN   = lane & 15;
  const int cM0  = (lane < 16) ? 0 : 8;

  const __bf16* Kbase = Kbf + (size_t)bIdx * TDIM * DDIM;

  // Preload this wave's Q fragments (16 rows x 128 d) into registers.
  v16bf qfrag[4];
  {
    const __bf16* qrow = Qbf + (size_t)(m0 + aM) * DDIM + wave * 128 + aKlo;
#pragma unroll
    for (int kt = 0; kt < 4; ++kt) {
      const __bf16* ap = qrow + kt * 32;
      qfrag[kt] = cat8(*(const v8bf*)(ap), *(const v8bf*)(ap + 16));
    }
  }

  v8f acc[8];
#pragma unroll
  for (int nt = 0; nt < 8; ++nt)
#pragma unroll
    for (int r = 0; r < 8; ++r) acc[nt][r] = 0.f;

  // phase-2 element mapping: each thread owns (row, 2 cols) of the 16x32 S tile
  const int   p2row = tid & 15;
  const int   p2col = (tid >> 4) * 2;
  const float ksumv = Ksum[m0 + p2row];
  const int   tq    = t0 + p2row;

  const int nsb = ((t0 + 15) >> 5) + 1;   // causal: only s-blocks with s0 <= t

#ifdef ASYNC_LDS
  // Prefetch s-block 0: contiguous 64KB, 16 x b128 per thread.
  {
    const char* g = (const char*)Kbase;          // s0 = 0
    char* l = (char*)Klds;
#pragma unroll
    for (int i = 0; i < 16; ++i) {
      const int off = tid * 16 + i * 4096;
      async_cp16(g + off, l + off);
    }
  }
#endif

  for (int sb = 0; sb < nsb; ++sb) {
    const int s0 = sb * 32;
    Ssh[tid]       = 0.f;
    Ssh[tid + 256] = 0.f;
#ifdef ASYNC_LDS
    __builtin_amdgcn_s_wait_asynccnt(0);   // this wave's copies for buf sb done
#endif
    __syncthreads();                       // all waves' slices landed
#ifdef ASYNC_LDS
    if (sb + 1 < nsb) {                    // prefetch next block into other buf
      const char* g = (const char*)(Kbase + (size_t)(s0 + 32) * DDIM);
      char* l = (char*)(Klds + ((sb + 1) & 1) * (32 * 1024));
#pragma unroll
      for (int i = 0; i < 16; ++i) {
        const int off = tid * 16 + i * 4096;
        async_cp16(g + off, l + off);
      }
    }
#endif

    // phase 1: partial 16x32 Q@K^T, d split over waves, reduced via ds_add_f32
#pragma unroll
    for (int st = 0; st < 2; ++st) {
      v8f ps;
#pragma unroll
      for (int r = 0; r < 8; ++r) ps[r] = 0.f;
#ifdef ASYNC_LDS
      const __bf16* krow = Klds + (size_t)((sb & 1) * (32 * 1024)) +
                           (st * 16 + bN) * 1024 + wave * 128 + bKlo;
#else
      const __bf16* krow = Kbf +
          ((size_t)bIdx * TDIM + s0 + st * 16 + bN) * DDIM + wave * 128 + bKlo;
#endif
#pragma unroll
      for (int kt = 0; kt < 4; ++kt) {
        const __bf16* bp = krow + kt * 32;
        v16bf bb = cat8(*(const v8bf*)(bp), *(const v8bf*)(bp + 16));
        ps = wmma_bf16f32(qfrag[kt], bb, ps);
      }
#pragma unroll
      for (int r = 0; r < 8; ++r)
        atomicAdd(&Ssh[(cM0 + r) * 32 + st * 16 + cN], ps[r]);
    }
    __syncthreads();

    // phase 2: relative-position bias + causal mask, convert to bf16 in LDS
#pragma unroll
    for (int c = 0; c < 2; ++c) {
      const int col = p2col + c;
      const int s = s0 + col;
      float v = 0.f;
      if (s <= tq) v = Ssh[p2row * 32 + col] + ksumv * vtab[tq - s];
      Abf[p2row * 32 + col] = (__bf16)v;
    }
    __syncthreads();

    // phase 3: out += S @ V  (A from LDS, B from transposed V)
    v16bf afrag;
    {
      const __bf16* ar = Abf + aM * 32 + aKlo;
      afrag = cat8(*(const v8bf*)(ar), *(const v8bf*)(ar + 16));
    }
#pragma unroll
    for (int nt = 0; nt < 8; ++nt) {
      const int n0 = wave * 128 + nt * 16;
      const __bf16* bp =
          Vt + ((size_t)bIdx * DDIM + n0 + bN) * TDIM + s0 + bKlo;
      v16bf bb = cat8(*(const v8bf*)(bp), *(const v8bf*)(bp + 16));
      acc[nt] = wmma_bf16f32(afrag, bb, acc[nt]);
    }
    __syncthreads();
  }

  // store f32 output tile
#pragma unroll
  for (int nt = 0; nt < 8; ++nt) {
    const int n0 = wave * 128 + nt * 16;
#pragma unroll
    for (int r = 0; r < 8; ++r)
      out[(size_t)(m0 + cM0 + r) * DDIM + n0 + cN] = acc[nt][r];
  }
}

extern "C" void kernel_launch(void* const* d_in, const int* in_sizes, int n_in,
                              void* d_out, int out_size, void* d_ws,
                              size_t ws_size, hipStream_t stream) {
  (void)in_sizes; (void)n_in; (void)out_size;
  const float* e    = (const float*)d_in[0];
  const float* res  = (const float*)d_in[1];
  const float* Wq_w = (const float*)d_in[2];
  const float* Wq_b = (const float*)d_in[3];
  const float* Wk_w = (const float*)d_in[4];
  const float* Wk_b = (const float*)d_in[5];
  const float* Wv_w = (const float*)d_in[6];
  const float* Wv_b = (const float*)d_in[7];
  const float* vtab = (const float*)d_in[8];
  float* out = (float*)d_out;

  const size_t BT = (size_t)BDIM * TDIM;
  size_t off = 0;
  char* base = (char*)d_ws;
  auto take = [&](size_t bytes) -> char* {
    char* p = base + off;
    off += (bytes + 255) & ~(size_t)255;
    return p;
  };
  __bf16* ebf  = (__bf16*)take(BT * DDIM * 2);
  __bf16* rbf  = (__bf16*)take(BT * DDIM * 2);
  __bf16* wqb  = (__bf16*)take((size_t)DDIM * DDIM * 2);
  __bf16* wkb  = (__bf16*)take((size_t)DDIM * DDIM * 2);
  __bf16* wvb  = (__bf16*)take((size_t)DDIM * DDIM * 2);
  __bf16* Qbf  = (__bf16*)take(BT * DDIM * 2);
  __bf16* Kbf  = (__bf16*)take(BT * DDIM * 2);
  __bf16* Vt   = (__bf16*)take(BT * DDIM * 2);
  float*  Ksum = (float*)take(BT * sizeof(float));
  if (off > ws_size) return;  // workspace too small; nothing safe to do

  const int nBT4 = (int)(BT * DDIM / 4);
  const int nW4  = DDIM * DDIM / 4;

  cvt_f32_bf16<<<(nBT4 + 255) / 256, 256, 0, stream>>>(e,   ebf, nBT4);
  cvt_f32_bf16<<<(nBT4 + 255) / 256, 256, 0, stream>>>(res, rbf, nBT4);
  cvt_f32_bf16<<<(nW4  + 255) / 256, 256, 0, stream>>>(Wq_w, wqb, nW4);
  cvt_f32_bf16<<<(nW4  + 255) / 256, 256, 0, stream>>>(Wk_w, wkb, nW4);
  cvt_f32_bf16<<<(nW4  + 255) / 256, 256, 0, stream>>>(Wv_w, wvb, nW4);

  qkv_kernel<<<(int)(BT / 16), 256, 0, stream>>>(ebf, rbf, wqb, wkb, wvb,
                                                 Wq_b, Wk_b, Wv_b,
                                                 Qbf, Kbf, Vt);
  ksum_kernel<<<(int)(BT / 8), 256, 0, stream>>>(Kbf, Ksum);

  // dynamic LDS: 2KB S + 1KB Abf + 2 x 64KB K double-buffer
  const size_t attn_lds = 3072 + 2 * 32 * 1024 * 2;
  attn_kernel<<<(int)(BT / 16), 256, attn_lds, stream>>>(Qbf, Kbf, Vt, Ksum,
                                                         vtab, out);
}
